// CasualSelfAttention_4449586118663
// MI455X (gfx1250) — compile-verified, matrix operand order checked by reading, not run on hardware
//
#include <hip/hip_runtime.h>
#include <hip/hip_bf16.h>
#include <math.h>

// ---------------------------------------------------------------------------
// Causal self-attention (GPT block) for MI455X / gfx1250 (wave32, WMMA).
// B=4, T=2048, C=1024, H=16, D=64.
// Pipeline: f32->bf16 convert pass, then three WMMA kernels whose LDS tiles
// are staged with double-buffered GLOBAL_LOAD_ASYNC_TO_LDS_B128 DMA.
// ---------------------------------------------------------------------------

typedef __attribute__((ext_vector_type(16))) __bf16 v16bf;
typedef __attribute__((ext_vector_type(8)))  float  v8f;
typedef __attribute__((ext_vector_type(4)))  int    v4i;

union FragBF { v16bf v; __bf16 e[16]; };
union AccF   { v8f   v; float  e[8];  };

static constexpr int BB  = 4;
static constexpr int TT  = 2048;
static constexpr int CC  = 1024;
static constexpr int HH  = 16;
static constexpr int DD  = 64;
static constexpr int BT  = BB * TT;   // 8192
static constexpr int C3  = 3 * CC;    // 3072

static constexpr int LDT = 72;        // bf16 elems; 144B rows (16B aligned)
static constexpr int BK  = 64;        // K-depth per LDS stage (2 WMMA k-steps)

// ---------------------------------------------------------------------------
// Async global->LDS DMA (CDNA5). Falls back to synchronous copy if the
// builtin is unavailable on this toolchain.
// ---------------------------------------------------------------------------
#if __has_builtin(__builtin_amdgcn_global_load_async_to_lds_b128)
#define HAVE_ASYNC 1
#else
#define HAVE_ASYNC 0
#endif

typedef __attribute__((address_space(1))) v4i gv4i;   // global int4
typedef __attribute__((address_space(3))) v4i lv4i;   // LDS int4

__device__ __forceinline__ void async_copy16(const void* g, void* l) {
#if HAVE_ASYNC
    __builtin_amdgcn_global_load_async_to_lds_b128((gv4i*)(void*)g, (lv4i*)l, 0, 0);
#else
    *(uint4*)l = *(const uint4*)g;
#endif
}

__device__ __forceinline__ void wait_async() {
#if HAVE_ASYNC
#if __has_builtin(__builtin_amdgcn_s_wait_asynccnt)
    __builtin_amdgcn_s_wait_asynccnt(0);
#else
    asm volatile("s_wait_asynccnt 0x0" ::: "memory");
#endif
#endif
}

// ---------------------------------------------------------------------------
// Fragment layout helpers (ISA 7.12.2, 16-bit A/B matrix 16x32):
// lanes 0-15 hold K={0..7,16..23}, lanes 16-31 hold K={8..15,24..31}.
// ---------------------------------------------------------------------------
__device__ __forceinline__ int a_k_of(int e, int half) {
    int i = e >> 1;
    int base = (i < 4) ? (2 * i) : (16 + 2 * (i - 4));
    return base + 8 * half + (e & 1);
}

// element (row = lane%16, k) from row-major tile p[row*ld + k]
__device__ __forceinline__ v16bf load_frag(const __bf16* p, int ld) {
    int lane = threadIdx.x & 31;
    int r = lane & 15, half = lane >> 4;
    FragBF f;
#pragma unroll
    for (int e = 0; e < 16; ++e) f.e[e] = p[r * ld + a_k_of(e, half)];
    return f.v;
}

// element (k, col = lane%16) from row-major tile p[k*ld + col]
__device__ __forceinline__ v16bf load_frag_t(const __bf16* p, int ld) {
    int lane = threadIdx.x & 31;
    int c = lane & 15, half = lane >> 4;
    FragBF f;
#pragma unroll
    for (int e = 0; e < 16; ++e) f.e[e] = p[a_k_of(e, half) * ld + c];
    return f.v;
}

#define WMMA_BF16(A, B, Cacc) \
    __builtin_amdgcn_wmma_f32_16x16x32_bf16(false, (A), false, (B), (short)0, (Cacc), false, false)

// ---------------------------------------------------------------------------
// Kernel 0: f32 -> bf16 bulk convert (vectorized)
// ---------------------------------------------------------------------------
__global__ __launch_bounds__(256)
void cvt_bf16_kernel(const float* __restrict__ src, __bf16* __restrict__ dst, int n4) {
    int i = blockIdx.x * 256 + threadIdx.x;
    if (i < n4) {
        float4 v = ((const float4*)src)[i];
        __bf16* d = dst + (size_t)i * 4;
        d[0] = (__bf16)v.x; d[1] = (__bf16)v.y;
        d[2] = (__bf16)v.z; d[3] = (__bf16)v.w;
    }
}

// ---------------------------------------------------------------------------
// Kernel 1: qkv = x @ W_attn^T + b_attn ; scatter to Q/K/V [B,H,T,D] bf16
// A = xb [BT][CC] bf16, B = Wab [C3][CC] bf16. 128x128 block, 8 waves.
// ---------------------------------------------------------------------------
__global__ __launch_bounds__(256)
void qkv_gemm_kernel(const __bf16* __restrict__ A, const __bf16* __restrict__ Wb,
                     const float* __restrict__ ba,
                     __bf16* __restrict__ Qb, __bf16* __restrict__ Kb,
                     __bf16* __restrict__ Vb)
{
    __shared__ __bf16 sA[2][128 * LDT];
    __shared__ __bf16 sB[2][128 * LDT];
    const int tid  = threadIdx.x;
    const int lane = tid & 31, wave = tid >> 5;
    const int wm = wave & 3, wn = wave >> 2;   // 4x2 wave grid -> 32x64 / wave
    const int m0 = blockIdx.x * 128;
    const int n0 = blockIdx.y * 128;

    AccF acc[2][4];
#pragma unroll
    for (int i = 0; i < 2; ++i)
#pragma unroll
        for (int j = 0; j < 4; ++j)
#pragma unroll
            for (int q = 0; q < 8; ++q) acc[i][j].e[q] = 0.f;

    auto stage = [&](int kt, int buf) {
        const int k0 = kt * BK;
#pragma unroll
        for (int r = 0; r < 4; ++r) {
            int c   = tid + r * 256;        // 16B chunk id, [0, 1024)
            int row = c >> 3;               // 8 chunks per 128B row
            int off = (c & 7) * 16;         // byte offset within row
            async_copy16((const char*)(A  + (size_t)(m0 + row) * CC + k0) + off,
                         (char*)&sA[buf][row * LDT] + off);
            async_copy16((const char*)(Wb + (size_t)(n0 + row) * CC + k0) + off,
                         (char*)&sB[buf][row * LDT] + off);
        }
    };

    const int NKT = CC / BK;   // 16 stages
    stage(0, 0);
    for (int kt = 0; kt < NKT; ++kt) {
        const int buf = kt & 1;
        wait_async();
        __syncthreads();
        if (kt + 1 < NKT) stage(kt + 1, buf ^ 1);
#pragma unroll
        for (int ks = 0; ks < 2; ++ks) {
            v16bf af[2], bfv[4];
#pragma unroll
            for (int mi = 0; mi < 2; ++mi)
                af[mi] = load_frag(&sA[buf][(wm * 32 + mi * 16) * LDT + ks * 32], LDT);
#pragma unroll
            for (int ni = 0; ni < 4; ++ni)
                bfv[ni] = load_frag(&sB[buf][(wn * 64 + ni * 16) * LDT + ks * 32], LDT);
#pragma unroll
            for (int mi = 0; mi < 2; ++mi)
#pragma unroll
                for (int ni = 0; ni < 4; ++ni)
                    acc[mi][ni].v = WMMA_BF16(af[mi], bfv[ni], acc[mi][ni].v);
        }
    }

    const int half = lane >> 4, nn = lane & 15;
#pragma unroll
    for (int mi = 0; mi < 2; ++mi)
#pragma unroll
        for (int ni = 0; ni < 4; ++ni)
#pragma unroll
            for (int r = 0; r < 8; ++r) {
                int gm = m0 + wm * 32 + mi * 16 + r + 8 * half;   // [0, 8192)
                int gn = n0 + wn * 64 + ni * 16 + nn;             // [0, 3072)
                float v = acc[mi][ni].e[r] + ba[gn];
                int b = gm >> 11, t = gm & (TT - 1);
                int sec = gn >> 10, c = gn & (CC - 1);
                int h = c >> 6, d = c & (DD - 1);
                size_t dst = (((size_t)b * HH + h) * TT + t) * DD + d;
                if      (sec == 0) Qb[dst] = (__bf16)v;
                else if (sec == 1) Kb[dst] = (__bf16)v;
                else               Vb[dst] = (__bf16)v;
            }
}

// ---------------------------------------------------------------------------
// Kernel 2: flash attention per (b, h, 128-query tile); 8 waves x 16 queries.
// K/V tiles double-buffered via async DMA; online softmax in registers.
// ---------------------------------------------------------------------------
__global__ __launch_bounds__(256)
void attn_kernel(const __bf16* __restrict__ Qb, const __bf16* __restrict__ Kb,
                 const __bf16* __restrict__ Vb, __bf16* __restrict__ Y)
{
    __shared__ __bf16 sK[2][32 * 64];
    __shared__ __bf16 sV[2][32 * 64];
    __shared__ __bf16 sP[8][16 * 32];   // per-wave P scratch (C-layout -> A-layout)

    const int tid  = threadIdx.x;
    const int lane = tid & 31, wave = tid >> 5;
    const int half = lane >> 4, nn = lane & 15;
    const int qt = blockIdx.x, h = blockIdx.y, b = blockIdx.z;
    const int q0_block = qt * 128;
    const int q0w = q0_block + wave * 16;
    const size_t base = ((size_t)(b * HH + h)) * TT * DD;

    // Q A-fragments (16 queries x 64 dims -> two 16x32 bf16 frags) from global.
    FragBF qf0, qf1;
#pragma unroll
    for (int e = 0; e < 16; ++e) {
        int k = a_k_of(e, half);
        qf0.e[e] = Qb[base + (size_t)(q0w + nn) * DD + k];
        qf1.e[e] = Qb[base + (size_t)(q0w + nn) * DD + 32 + k];
    }

    AccF o[4];
    float mrow[8], lrow[8];
#pragma unroll
    for (int dt = 0; dt < 4; ++dt)
#pragma unroll
        for (int q = 0; q < 8; ++q) o[dt].e[q] = 0.f;
#pragma unroll
    for (int r = 0; r < 8; ++r) { mrow[r] = -INFINITY; lrow[r] = 0.f; }

    auto stage_kv = [&](int kt, int buf) {
        const int kb = kt * 32;
        // 32x64 bf16 tile = 4096B; 256 lanes x 16B = whole tile per instruction.
        async_copy16((const char*)(Kb + base + (size_t)kb * DD) + tid * 16,
                     (char*)&sK[buf][0] + tid * 16);
        async_copy16((const char*)(Vb + base + (size_t)kb * DD) + tid * 16,
                     (char*)&sV[buf][0] + tid * 16);
    };

    const int nkt = (q0_block + 128) / 32;   // uniform across the block
    stage_kv(0, 0);
    for (int kt = 0; kt < nkt; ++kt) {
        const int kb  = kt * 32;
        const int buf = kt & 1;
        wait_async();
        __syncthreads();
        if (kt + 1 < nkt) stage_kv(kt + 1, buf ^ 1);

        if (kb <= q0w + 15) {   // causal: this wave has live keys in the tile
            AccF s0, s1;
#pragma unroll
            for (int q = 0; q < 8; ++q) { s0.e[q] = 0.f; s1.e[q] = 0.f; }

            // S = Q @ K^T : B-frag element (k=d, n=key) = sK[key*64 + d]
            v16bf bk0lo = load_frag(&sK[buf][0],            64);
            v16bf bk0hi = load_frag(&sK[buf][32],           64);
            v16bf bk1lo = load_frag(&sK[buf][16 * 64],      64);
            v16bf bk1hi = load_frag(&sK[buf][16 * 64 + 32], 64);
            s0.v = WMMA_BF16(qf0.v, bk0lo, s0.v);
            s0.v = WMMA_BF16(qf1.v, bk0hi, s0.v);
            s1.v = WMMA_BF16(qf0.v, bk1lo, s1.v);
            s1.v = WMMA_BF16(qf1.v, bk1hi, s1.v);

            const float sc = 0.125f;   // 1/sqrt(64)
#pragma unroll
            for (int r = 0; r < 8; ++r) {
                int qg = q0w + r + 8 * half;
                float a0 = s0.e[r] * sc;
                float a1 = s1.e[r] * sc;
                if (kb + nn      > qg) a0 = -INFINITY;
                if (kb + 16 + nn > qg) a1 = -INFINITY;
                float rm = fmaxf(a0, a1);
#pragma unroll
                for (int off = 8; off > 0; off >>= 1)
                    rm = fmaxf(rm, __shfl_xor(rm, off, 16));
                float mnew  = fmaxf(mrow[r], rm);
                float scale = __expf(mrow[r] - mnew);
                mrow[r] = mnew;
                float p0 = __expf(a0 - mnew);
                float p1 = __expf(a1 - mnew);
                float rs = p0 + p1;
#pragma unroll
                for (int off = 8; off > 0; off >>= 1)
                    rs += __shfl_xor(rs, off, 16);
                lrow[r] = lrow[r] * scale + rs;
#pragma unroll
                for (int dt = 0; dt < 4; ++dt) o[dt].e[r] *= scale;
                int m = r + 8 * half;
                sP[wave][m * 32 + nn]      = (__bf16)p0;
                sP[wave][m * 32 + 16 + nn] = (__bf16)p1;
            }

            // O += P @ V : A = P (16x32), B element (k=key, n=d) = sV[key*64+d]
            v16bf pf = load_frag(&sP[wave][0], 32);
#pragma unroll
            for (int dt = 0; dt < 4; ++dt) {
                v16bf vf = load_frag_t(&sV[buf][dt * 16], 64);
                o[dt].v = WMMA_BF16(pf, vf, o[dt].v);
            }
        }
        __syncthreads();
    }

    // Normalize and write y[b, q, h*64 + d] (bf16, consumed by proj GEMM)
#pragma unroll
    for (int r = 0; r < 8; ++r) {
        int qg = q0w + r + 8 * half;
        float inv = 1.0f / lrow[r];
#pragma unroll
        for (int dt = 0; dt < 4; ++dt) {
            int col = h * DD + dt * 16 + nn;
            Y[((size_t)b * TT + qg) * CC + col] = (__bf16)(o[dt].e[r] * inv);
        }
    }
}

// ---------------------------------------------------------------------------
// Kernel 3: out = Y @ W_proj^T + b_proj   (bf16 in, f32 out)
// ---------------------------------------------------------------------------
__global__ __launch_bounds__(256)
void proj_gemm_kernel(const __bf16* __restrict__ A, const __bf16* __restrict__ Wb,
                      const float* __restrict__ bp, float* __restrict__ out)
{
    __shared__ __bf16 sA[2][128 * LDT];
    __shared__ __bf16 sB[2][128 * LDT];
    const int tid  = threadIdx.x;
    const int lane = tid & 31, wave = tid >> 5;
    const int wm = wave & 3, wn = wave >> 2;
    const int m0 = blockIdx.x * 128;
    const int n0 = blockIdx.y * 128;

    AccF acc[2][4];
#pragma unroll
    for (int i = 0; i < 2; ++i)
#pragma unroll
        for (int j = 0; j < 4; ++j)
#pragma unroll
            for (int q = 0; q < 8; ++q) acc[i][j].e[q] = 0.f;

    auto stage = [&](int kt, int buf) {
        const int k0 = kt * BK;
#pragma unroll
        for (int r = 0; r < 4; ++r) {
            int c   = tid + r * 256;
            int row = c >> 3;
            int off = (c & 7) * 16;
            async_copy16((const char*)(A  + (size_t)(m0 + row) * CC + k0) + off,
                         (char*)&sA[buf][row * LDT] + off);
            async_copy16((const char*)(Wb + (size_t)(n0 + row) * CC + k0) + off,
                         (char*)&sB[buf][row * LDT] + off);
        }
    };

    const int NKT = CC / BK;
    stage(0, 0);
    for (int kt = 0; kt < NKT; ++kt) {
        const int buf = kt & 1;
        wait_async();
        __syncthreads();
        if (kt + 1 < NKT) stage(kt + 1, buf ^ 1);
#pragma unroll
        for (int ks = 0; ks < 2; ++ks) {
            v16bf af[2], bfv[4];
#pragma unroll
            for (int mi = 0; mi < 2; ++mi)
                af[mi] = load_frag(&sA[buf][(wm * 32 + mi * 16) * LDT + ks * 32], LDT);
#pragma unroll
            for (int ni = 0; ni < 4; ++ni)
                bfv[ni] = load_frag(&sB[buf][(wn * 64 + ni * 16) * LDT + ks * 32], LDT);
#pragma unroll
            for (int mi = 0; mi < 2; ++mi)
#pragma unroll
                for (int ni = 0; ni < 4; ++ni)
                    acc[mi][ni].v = WMMA_BF16(af[mi], bfv[ni], acc[mi][ni].v);
        }
    }

    const int half = lane >> 4, nn = lane & 15;
#pragma unroll
    for (int mi = 0; mi < 2; ++mi)
#pragma unroll
        for (int ni = 0; ni < 4; ++ni)
#pragma unroll
            for (int r = 0; r < 8; ++r) {
                int gm = m0 + wm * 32 + mi * 16 + r + 8 * half;
                int gn = n0 + wn * 64 + ni * 16 + nn;
                out[(size_t)gm * CC + gn] = acc[mi][ni].e[r] + bp[gn];
            }
}

// ---------------------------------------------------------------------------
extern "C" void kernel_launch(void* const* d_in, const int* in_sizes, int n_in,
                              void* d_out, int out_size, void* d_ws, size_t ws_size,
                              hipStream_t stream) {
    (void)in_sizes; (void)n_in; (void)out_size; (void)ws_size;
    const float* x  = (const float*)d_in[0];
    const float* Wa = (const float*)d_in[1];
    const float* ba = (const float*)d_in[2];
    const float* Wp = (const float*)d_in[3];
    const float* bp = (const float*)d_in[4];
    float* out = (float*)d_out;

    // Workspace: xb | Wab | Wpb | Q | K | V | Y  (all bf16)
    const size_t nX   = (size_t)BT * CC;        // 8Mi
    const size_t nWa  = (size_t)C3 * CC;        // 3Mi
    const size_t nWp  = (size_t)CC * CC;        // 1Mi
    const size_t nQKV = (size_t)BB * HH * TT * DD;  // 8Mi each
    __bf16* xb  = (__bf16*)d_ws;
    __bf16* Wab = xb  + nX;
    __bf16* Wpb = Wab + nWa;
    __bf16* Qb  = Wpb + nWp;
    __bf16* Kb  = Qb  + nQKV;
    __bf16* Vb  = Kb  + nQKV;
    __bf16* Yb  = Vb  + nQKV;

    cvt_bf16_kernel<<<(int)(nX  / 4 + 255) / 256, 256, 0, stream>>>(x,  xb,  (int)(nX  / 4));
    cvt_bf16_kernel<<<(int)(nWa / 4 + 255) / 256, 256, 0, stream>>>(Wa, Wab, (int)(nWa / 4));
    cvt_bf16_kernel<<<(int)(nWp / 4 + 255) / 256, 256, 0, stream>>>(Wp, Wpb, (int)(nWp / 4));

    dim3 g1(BT / 128, C3 / 128);
    qkv_gemm_kernel<<<g1, 256, 0, stream>>>(xb, Wab, ba, Qb, Kb, Vb);

    dim3 g2(TT / 128, HH, BB);
    attn_kernel<<<g2, 256, 0, stream>>>(Qb, Kb, Vb, Yb);

    dim3 g3(BT / 128, CC / 128);
    proj_gemm_kernel<<<g3, 256, 0, stream>>>(Yb, Wpb, bp, out);
}
